// ScaledDotProductAttentionWithReferenceDrop_87634512708217
// MI455X (gfx1250) — matrix operand order repaired and checked by reading, not compiled, not run on hardware
//
#include <hip/hip_runtime.h>

// Flash-attention forward with reference column + sigmoid(relu) gate.
// CDNA5 (gfx1250), wave32, v_wmma_f32_16x16x32_f16.
// Q staged in LDS (f16); next K/V tile prefetched during compute;
// softmax row reductions done with DPP16 (VALU) instead of ds_bpermute.

typedef __attribute__((ext_vector_type(16))) _Float16     v16h;
typedef __attribute__((ext_vector_type(8)))  float        v8f;
typedef __attribute__((ext_vector_type(4)))  float        f32x4;
typedef __attribute__((ext_vector_type(4)))  unsigned int u32x4;
typedef __attribute__((ext_vector_type(2)))  unsigned int u32x2;

#define NBATCH 8
#define LQ     2048
#define LK     2048
#define DHEAD  256
#define DVAL   256

#define BLK_M  64     // queries per workgroup
#define NWAVE  4      // 128 threads
#define TK     32     // keys per iteration (WMMA K for the P*V gemm)

#define QPITCH 264    // 256 + 8 halfs pad
#define KPITCH 264    // 256 + 8 halfs pad
#define VPITCH 40     // 32 keys + 8 halfs pad (transposed V)
#define PPITCH 40     // 32 cols + 8 halfs pad

union Frag {
  v16h         v;
  _Float16     h[16];
  u32x4        q[2];
};

static __device__ __forceinline__ unsigned short f2h_bits(float f) {
  _Float16 h = (_Float16)f;
  return __builtin_bit_cast(unsigned short, h);
}

// DPP16 cross-lane permute (stays in the VALU, off the LDS pipe).
// CTRL must be a compile-time constant -> template parameter.
template <int CTRL>
static __device__ __forceinline__ float dpp_bf(float x) {
  return __builtin_bit_cast(float,
      __builtin_amdgcn_mov_dpp(__builtin_bit_cast(int, x), CTRL, 0xf, 0xf, true));
}
// butterfly reduction over each 16-lane row:
// quad_perm xor1 (0xB1), quad_perm xor2 (0x4E), row_half_mirror (0x141), row_mirror (0x140)
static __device__ __forceinline__ float rowmax16(float x) {
  x = fmaxf(x, dpp_bf<0xB1>(x));
  x = fmaxf(x, dpp_bf<0x4E>(x));
  x = fmaxf(x, dpp_bf<0x141>(x));
  x = fmaxf(x, dpp_bf<0x140>(x));
  return x;
}
static __device__ __forceinline__ float rowsum16(float x) {
  x += dpp_bf<0xB1>(x);
  x += dpp_bf<0x4E>(x);
  x += dpp_bf<0x141>(x);
  x += dpp_bf<0x140>(x);
  return x;
}

__global__ __launch_bounds__(128)
void fa_refdrop_kernel(const float* __restrict__ Q,
                       const float* __restrict__ K,
                       const float* __restrict__ V,
                       const int*   __restrict__ MASK,
                       float* __restrict__ O)
{
  __shared__ unsigned short Qs[BLK_M * QPITCH];       // Q tile, row-major (q x d), f16
  __shared__ unsigned short Ks[TK * KPITCH];          // K tile, row-major (key x d), f16
  __shared__ unsigned short Vs[DVAL * VPITCH];        // V tile, transposed (dv x key), f16
  __shared__ unsigned short Ps[NWAVE * 16 * PPITCH];  // per-wave P staging (q x key), f16

  const int tid  = threadIdx.x;
  const int lane = tid & 31;
  const int wid  = tid >> 5;
  const int l16  = lane & 15;
  const int hi   = lane >> 4;        // 0 for lanes 0-15, 1 for lanes 16-31

  const int b     = blockIdx.y;
  const int qbase = blockIdx.x * BLK_M;
  const int wq    = wid * 16;        // this wave's query row tile inside the block

  // ---- cooperative stage: Q tile (64 x 256) fp32 -> f16 row-major in LDS ----
  {
    const float* qg = Q + ((size_t)b * LQ + qbase) * DHEAD;
    #pragma unroll
    for (int i = 0; i < 32; ++i) {
      int lin = i * 128 + tid;        // 0..4095 -> 64 rows x 64 float4 cols
      int row = lin >> 6;             // q row 0..63
      int c4  = lin & 63;             // float4 column
      f32x4 x = *(const f32x4*)(qg + row * DHEAD + c4 * 4);
      u32x2 p;
      p.x = (unsigned)f2h_bits(x.x) | ((unsigned)f2h_bits(x.y) << 16);
      p.y = (unsigned)f2h_bits(x.z) | ((unsigned)f2h_bits(x.w) << 16);
      *(u32x2*)(&Qs[row * QPITCH + c4 * 4]) = p;
    }
  }
  // visibility of Qs is guaranteed by the barrier at the top of iteration 0

  // ---- running state: 16x256 f32 accumulator, row max m, row sum l ----
  v8f acc[16];
  v8f mrow, lrow;
  #pragma unroll
  for (int j = 0; j < 16; ++j)
    #pragma unroll
    for (int r = 0; r < 8; ++r) acc[j][r] = 0.0f;
  #pragma unroll
  for (int r = 0; r < 8; ++r) { mrow[r] = -1.0e30f; lrow[r] = 0.0f; }

  unsigned short* pw = &Ps[wid * 16 * PPITCH];

  for (int kb = 0; kb < LK; kb += TK) {
    __syncthreads();   // protect LDS tiles from previous iteration's readers

    // ---- cooperative stage: K tile row-major, V tile transposed, fp32 -> f16 ----
    const float* kg = K + ((size_t)b * LK + kb) * DHEAD;
    const float* vg = V + ((size_t)b * LK + kb) * DVAL;
    #pragma unroll
    for (int i = 0; i < 16; ++i) {
      int lin = i * 128 + tid;        // 0..2047 -> 32 rows x 64 float4 cols
      int row = lin >> 6;             // key row 0..31
      int c4  = lin & 63;             // float4 column

      f32x4 xk = *(const f32x4*)(kg + row * DHEAD + c4 * 4);
      u32x2 pk;
      pk.x = (unsigned)f2h_bits(xk.x) | ((unsigned)f2h_bits(xk.y) << 16);
      pk.y = (unsigned)f2h_bits(xk.z) | ((unsigned)f2h_bits(xk.w) << 16);
      *(u32x2*)(&Ks[row * KPITCH + c4 * 4]) = pk;

      f32x4 xv = *(const f32x4*)(vg + row * DVAL + c4 * 4);
      Vs[(c4 * 4 + 0) * VPITCH + row] = f2h_bits(xv.x);
      Vs[(c4 * 4 + 1) * VPITCH + row] = f2h_bits(xv.y);
      Vs[(c4 * 4 + 2) * VPITCH + row] = f2h_bits(xv.z);
      Vs[(c4 * 4 + 3) * VPITCH + row] = f2h_bits(xv.w);
    }
    __syncthreads();

    // ---- prefetch next K/V tile into cache while we compute this one ----
    if (kb + TK < LK) {
      const char* kn = (const char*)(K + ((size_t)b * LK + kb + TK) * DHEAD);
      const char* vn = (const char*)(V + ((size_t)b * LK + kb + TK) * DVAL);
      // 32KB per tensor tile / 128 threads = 256B = 2 cache lines per thread
      __builtin_prefetch(kn + tid * 256,       0, 3);
      __builtin_prefetch(kn + tid * 256 + 128, 0, 3);
      __builtin_prefetch(vn + tid * 256,       0, 3);
      __builtin_prefetch(vn + tid * 256 + 128, 0, 3);
    }

    // additive mask terms for the two 16-key column tiles (column = l16 in C layout)
    const float madd0 = MASK[(size_t)b * LK + kb + l16]      ? 0.0f : -1.0e9f;
    const float madd1 = MASK[(size_t)b * LK + kb + 16 + l16] ? 0.0f : -1.0e9f;

    // ---- S = Q K^T : two 16x16 f32 tiles, K-dim = D = 8 x 32 ----
    v8f s0, s1;
    #pragma unroll
    for (int r = 0; r < 8; ++r) { s0[r] = 0.0f; s1[r] = 0.0f; }
    #pragma unroll
    for (int c = 0; c < 8; ++c) {
      // A fragment (Q): lanes 0-15 hold K=0..7 & 16..23, hi lanes 8..15 & 24..31
      Frag qa;
      const int arow = wq + l16;
      const int ac0  = 32 * c + hi * 8;
      qa.q[0] = *(const u32x4*)(&Qs[arow * QPITCH + ac0]);
      qa.q[1] = *(const u32x4*)(&Qs[arow * QPITCH + ac0 + 16]);

      // B fragments (K^T): lanes 0-15 hold K=0..7 & 16..23, hi lanes 8..15 & 24..31
      const int kc0 = 32 * c + hi * 8;
      Frag kf0, kf1;
      kf0.q[0] = *(const u32x4*)(&Ks[l16 * KPITCH + kc0]);
      kf0.q[1] = *(const u32x4*)(&Ks[l16 * KPITCH + kc0 + 16]);
      kf1.q[0] = *(const u32x4*)(&Ks[(16 + l16) * KPITCH + kc0]);
      kf1.q[1] = *(const u32x4*)(&Ks[(16 + l16) * KPITCH + kc0 + 16]);

      s0 = __builtin_amdgcn_wmma_f32_16x16x32_f16(false, qa.v, false, kf0.v,
                                                  (short)0, s0, false, false);
      s1 = __builtin_amdgcn_wmma_f32_16x16x32_f16(false, qa.v, false, kf1.v,
                                                  (short)0, s1, false, false);
    }

    // scale by 1/sqrt(256) and apply additive mask
    #pragma unroll
    for (int r = 0; r < 8; ++r) {
      s0[r] = s0[r] * 0.0625f + madd0;
      s1[r] = s1[r] * 0.0625f + madd1;
    }

    // ---- online softmax update (row lives across a 16-lane half in each VGPR) ----
    v8f mnew, alpha;
    #pragma unroll
    for (int r = 0; r < 8; ++r) {
      float x  = rowmax16(fmaxf(s0[r], s1[r]));
      float mn = fmaxf(mrow[r], x);
      mnew[r]  = mn;
      alpha[r] = __expf(mrow[r] - mn);
    }
    #pragma unroll
    for (int j = 0; j < 16; ++j) acc[j] *= alpha;

    // p = exp(s-m); gate = sigmoid(relu(s)); numerator uses p*g, denominator uses p only
    v8f psum;
    #pragma unroll
    for (int r = 0; r < 8; ++r) {
      float p0 = __expf(s0[r] - mnew[r]);
      float p1 = __expf(s1[r] - mnew[r]);
      float g0 = 1.0f / (1.0f + __expf(-fmaxf(s0[r], 0.0f)));
      float g1 = 1.0f / (1.0f + __expf(-fmaxf(s1[r], 0.0f)));
      pw[(r + 8 * hi) * PPITCH + l16]      = f2h_bits(p0 * g0);
      pw[(r + 8 * hi) * PPITCH + 16 + l16] = f2h_bits(p1 * g1);
      psum[r] = p0 + p1;
    }
    #pragma unroll
    for (int r = 0; r < 8; ++r) {
      float x = rowsum16(psum[r]);
      lrow[r] = lrow[r] * alpha[r] + x;
      mrow[r] = mnew[r];
    }

    // ---- P fragment (A layout) from per-wave LDS staging (same-wave LDS ops in-order) ----
    Frag pa;
    {
      const int pc0 = hi * 8;
      pa.q[0] = *(const u32x4*)(&pw[l16 * PPITCH + pc0]);
      pa.q[1] = *(const u32x4*)(&pw[l16 * PPITCH + pc0 + 16]);
    }

    // ---- acc += P * V : 16 n-tiles over Dv=256, K=32 keys ----
    #pragma unroll
    for (int j = 0; j < 16; ++j) {
      Frag vb;
      const int vrow = 16 * j + l16;   // dv column (row of transposed V tile)
      const int vc0  = hi * 8;         // key offset per lane half (keys 8h..8h+7, 16+8h..)
      vb.q[0] = *(const u32x4*)(&Vs[vrow * VPITCH + vc0]);
      vb.q[1] = *(const u32x4*)(&Vs[vrow * VPITCH + vc0 + 16]);
      acc[j] = __builtin_amdgcn_wmma_f32_16x16x32_f16(false, pa.v, false, vb.v,
                                                      (short)0, acc[j], false, false);
    }
  }

  // ---- reference logit column: logit 0, zero value row -> only adds exp(0 - m) to denom ----
  v8f rinv;
  #pragma unroll
  for (int r = 0; r < 8; ++r) {
    float l = lrow[r] + __expf(-mrow[r]);
    rinv[r] = 1.0f / l;
  }

  // ---- write output (C layout: row = r + 8*hi, col = 16*j + l16) ----
  float* og = O + ((size_t)b * LQ + qbase + wq) * DVAL;
  #pragma unroll
  for (int j = 0; j < 16; ++j)
    #pragma unroll
    for (int r = 0; r < 8; ++r)
      og[(size_t)(r + 8 * hi) * DVAL + 16 * j + l16] = acc[j][r] * rinv[r];
}

extern "C" void kernel_launch(void* const* d_in, const int* in_sizes, int n_in,
                              void* d_out, int out_size, void* d_ws, size_t ws_size,
                              hipStream_t stream) {
  const float* q = (const float*)d_in[0];
  const float* k = (const float*)d_in[1];
  const float* v = (const float*)d_in[2];
  const int*   m = (const int*)d_in[3];
  float*       o = (float*)d_out;

  dim3 grid(LQ / BLK_M, NBATCH);   // 32 x 8 = 256 workgroups
  dim3 block(128);                 // 4 waves (wave32)
  hipLaunchKernelGGL(fa_refdrop_kernel, grid, block, 0, stream, q, k, v, m, o);
}
